// E74CUDADiagonalCell_23751169147418
// MI455X (gfx1250) — compile-verified, matrix operand order checked by reading, not run on hardware
//
#include <hip/hip_runtime.h>

// ---------------------------------------------------------------------------
// Problem dims (from reference): T=2048, B=16, D=1024, N=1024
// ---------------------------------------------------------------------------
#define T_STEPS 2048
#define BATCH   16
#define DDIM    1024
#define NDIM    1024
#define M_ROWS  (T_STEPS * BATCH)   // 32768 rows of x (flattened [t,b])
#define NTOT    (3 * NDIM)          // k|v|q weight rows concatenated: 3072

typedef __bf16 bf16_t;
typedef __bf16 v8bf  __attribute__((ext_vector_type(8)));
typedef __bf16 v16bf __attribute__((ext_vector_type(16)));
typedef float  v8f   __attribute__((ext_vector_type(8)));
typedef unsigned int u32x4 __attribute__((ext_vector_type(4)));
typedef int          i32x4 __attribute__((ext_vector_type(4)));
typedef int          i32x8 __attribute__((ext_vector_type(8)));

// workspace layout:
//   [0, 64MB)    x in bf16           (M_ROWS * DDIM ushorts)
//   [64, 70MB)   W_k|W_v|W_q bf16    (NTOT * DDIM ushorts)
//   [70, 454MB)  k,v,q in f32        (3 * M_ROWS * NDIM floats)
#define XB_ELEMS  ((size_t)M_ROWS * DDIM)
#define WB_ELEMS  ((size_t)NTOT * DDIM)

// GEMM staging
#define SLAB    64                 // K elements per TDM stage
#define STAGES  (DDIM / SLAB)      // 16
#define ROW_USH 72                 // LDS row stride in ushorts: 128B data + 16B TDM pad
#define TILE_USH (128 * ROW_USH)   // one 128 x SLAB bf16 tile in LDS

// ---------------------------------------------------------------------------
// f32 -> bf16 (round-to-nearest-even) conversion, vectorized 4-wide
// ---------------------------------------------------------------------------
__device__ __forceinline__ unsigned int bf16_rn(float f) {
    unsigned int u = __float_as_uint(f);
    u += 0x7fffu + ((u >> 16) & 1u);
    return u >> 16;
}

__global__ __launch_bounds__(256) void cvt_bf16_kernel(
    const float* __restrict__ src, unsigned short* __restrict__ dst, int n4) {
    int i = blockIdx.x * blockDim.x + threadIdx.x;
    int stride = gridDim.x * blockDim.x;
    const float4* s4 = (const float4*)src;
    uint2* d2 = (uint2*)dst;
    for (; i < n4; i += stride) {
        float4 f = s4[i];
        uint2 o;
        o.x = bf16_rn(f.x) | (bf16_rn(f.y) << 16);
        o.y = bf16_rn(f.z) | (bf16_rn(f.w) << 16);
        d2[i] = o;
    }
}

// ---------------------------------------------------------------------------
// Tensor Data Mover: 2D tile (tile_d1 rows x tile_d0 elems, bf16) -> LDS,
// with LDS padding: pad_interval=128B (code 4), pad_amount=16B (code 3),
// giving a 144B padded row stride (conflict-spread, 16B-aligned chunks).
// Descriptor layout per CDNA5 ISA.
// Toolchain here is the 6-arg builtin flavor (clang-23 / therock-10.0).
// ---------------------------------------------------------------------------
__device__ __forceinline__ void tdm_load_tile_2d(
    unsigned lds_byte_off, const void* gaddr,
    unsigned tensor_d0, unsigned tensor_d1, unsigned stride0,
    unsigned tile_d0, unsigned tile_d1) {
    unsigned long long ga = (unsigned long long)gaddr;
    u32x4 g0;
    g0.x = 1u;                                   // count=1, user descriptor
    g0.y = lds_byte_off;                         // lds_addr
    g0.z = (unsigned)ga;                         // global_addr[31:0]
    g0.w = (unsigned)(ga >> 32) | (2u << 30);    // global_addr[56:32] | type=2
    i32x8 g1;
    g1[0] = (int)((1u << 16)                     // data_size = 2 bytes
                | (1u << 20)                     // pad_enable
                | (4u << 22)                     // pad_interval: 128B
                | (3u << 25));                   // pad_amount: 4 DWORDs (16B)
    g1[1] = (int)((tensor_d0 & 0xFFFFu) << 16);            // dim0[15:0]
    g1[2] = (int)((tensor_d0 >> 16) | ((tensor_d1 & 0xFFFFu) << 16));
    g1[3] = (int)((tensor_d1 >> 16) | (tile_d0 << 16));    // dim1 hi | tile_dim0
    g1[4] = (int)(tile_d1 & 0xFFFFu);                      // tile_dim1, tile_dim2=0
    g1[5] = (int)stride0;                                  // dim0_stride[31:0]
    g1[6] = 0;                                             // stride0 hi | stride1 lo
    g1[7] = 0;                                             // stride1 hi
    i32x4 z4 = {0, 0, 0, 0};                               // groups 2/3 unused (2D)
    i32x8 z8 = {0, 0, 0, 0, 0, 0, 0, 0};
    __builtin_amdgcn_tensor_load_to_lds(g0, g1, z4, z4, z8, 0);
}

// ---------------------------------------------------------------------------
// GEMM: kvq[w][m][n] = sum_d x[m][d] * W_w[n][d]   (NT, both row-major in K)
// Block = 4 waves (128 threads), 128x128 block tile, 64x64 per wave.
// TDM double-buffered LDS staging; bf16 WMMA 16x16x32, f32 accumulate.
//
// A-fragment (16-bit A 16x32, ISA 7.12.2): lane L -> row L%16, K chunks
// [kk + 8*(L/16), +8) and [kk + 16 + 8*(L/16), +8): two 16B ds_load_b128.
// B mirrored with N = L%16.
// ---------------------------------------------------------------------------
__global__ __launch_bounds__(128) void gemm_bf16_kernel(
    const bf16_t* __restrict__ xb,   // [M_ROWS][DDIM]
    const bf16_t* __restrict__ wb,   // [NTOT][DDIM]
    float* __restrict__ kvq) {       // [3][M_ROWS][NDIM]
    __shared__ __align__(128) unsigned short smem[2][2][TILE_USH]; // [buf][A/B]

    const int lane = threadIdx.x & 31;
    const int wid  = threadIdx.x >> 5;
    const int r16  = lane & 15;   // row (A) / col (B/C) within 16x16 tile
    const int h    = lane >> 4;   // K-half selector

    const int bm0 = blockIdx.y * 128;
    const int bn0 = blockIdx.x * 128;
    const int wm  = (wid >> 1) * 64;   // wave offset inside block tile
    const int wn  = (wid & 1) * 64;

    const unsigned ldsA[2] = {(unsigned)(size_t)&smem[0][0][0],
                              (unsigned)(size_t)&smem[1][0][0]};
    const unsigned ldsB[2] = {(unsigned)(size_t)&smem[0][1][0],
                              (unsigned)(size_t)&smem[1][1][0]};

    v8f c[4][4];
    const v8f vzero = {0.f, 0.f, 0.f, 0.f, 0.f, 0.f, 0.f, 0.f};
#pragma unroll
    for (int i = 0; i < 4; ++i)
#pragma unroll
        for (int j = 0; j < 4; ++j) c[i][j] = vzero;

    // stage 0 prefetch
    if (wid == 0) {
        tdm_load_tile_2d(ldsA[0], xb + (size_t)bm0 * DDIM, DDIM, M_ROWS, DDIM,
                         SLAB, 128);
        tdm_load_tile_2d(ldsB[0], wb + (size_t)bn0 * DDIM, DDIM, NTOT, DDIM,
                         SLAB, 128);
    }

    int buf = 0;
    for (int s = 0; s < STAGES; ++s) {
        if (wid == 0) {
            if (s + 1 < STAGES) {
                const size_t kk = (size_t)(s + 1) * SLAB;
                tdm_load_tile_2d(ldsA[buf ^ 1], xb + (size_t)bm0 * DDIM + kk,
                                 DDIM, M_ROWS, DDIM, SLAB, 128);
                tdm_load_tile_2d(ldsB[buf ^ 1], wb + (size_t)bn0 * DDIM + kk,
                                 DDIM, NTOT, DDIM, SLAB, 128);
                __builtin_amdgcn_s_wait_tensorcnt(2);  // stage s complete
            } else {
                __builtin_amdgcn_s_wait_tensorcnt(0);
            }
        }
        __syncthreads();

        const unsigned short* at = &smem[buf][0][0];
        const unsigned short* bt = &smem[buf][1][0];
#pragma unroll
        for (int kk2 = 0; kk2 < SLAB; kk2 += 32) {
            const int sub = kk2 + h * 8;   // ushort offset inside padded row
            v16bf a[4], b[4];
#pragma unroll
            for (int i = 0; i < 4; ++i) {
                const unsigned short* p = at + (size_t)(wm + i * 16 + r16) * ROW_USH + sub;
                v8bf lo = *(const v8bf*)p;
                v8bf hi = *(const v8bf*)(p + 16);
                a[i] = __builtin_shufflevector(lo, hi, 0, 1, 2, 3, 4, 5, 6, 7,
                                                       8, 9, 10, 11, 12, 13, 14, 15);
            }
#pragma unroll
            for (int j = 0; j < 4; ++j) {
                const unsigned short* p = bt + (size_t)(wn + j * 16 + r16) * ROW_USH + sub;
                v8bf lo = *(const v8bf*)p;
                v8bf hi = *(const v8bf*)(p + 16);
                b[j] = __builtin_shufflevector(lo, hi, 0, 1, 2, 3, 4, 5, 6, 7,
                                                       8, 9, 10, 11, 12, 13, 14, 15);
            }
#pragma unroll
            for (int i = 0; i < 4; ++i)
#pragma unroll
                for (int j = 0; j < 4; ++j)
                    c[i][j] = __builtin_amdgcn_wmma_f32_16x16x32_bf16(
                        false, a[i], false, b[j], (short)0, c[i][j], false, false);
        }
        __syncthreads();   // all waves done reading buf before it is rewritten
        buf ^= 1;
    }

    // C layout (32-bit 16x16): lane L -> N = L%16; VGPR r -> M = r + 8*(L/16)
#pragma unroll
    for (int j = 0; j < 4; ++j) {
        const int ng = bn0 + wn + j * 16 + r16;   // global column in [0, 3072)
        const int w  = ng >> 10;                  // which of k/v/q
        const int nl = ng & 1023;
        float* op = kvq + ((size_t)w * M_ROWS) * NDIM + nl;
#pragma unroll
        for (int i = 0; i < 4; ++i) {
            const int mb = bm0 + wm + i * 16 + h * 8;
#pragma unroll
            for (int r = 0; r < 8; ++r)
                op[(size_t)(mb + r) * NDIM] = c[i][j][r];
        }
    }
}

// ---------------------------------------------------------------------------
// Diagonal scan: 16384 independent chains over T. 4 chains per thread
// (float4 / b128 streaming loads). out = [T,B,N] gated readout + [B,N] s_final
// ---------------------------------------------------------------------------
__global__ __launch_bounds__(128) void scan_kernel(
    const float* __restrict__ kvq,  // [3][M_ROWS][NDIM]
    const float* __restrict__ s0,   // [B][N]
    float* __restrict__ out) {      // [T][B][N] + [B][N]
    const int tid = blockIdx.x * blockDim.x + threadIdx.x;  // 0..4095
    const int e   = tid * 4;                                 // index in [B*N)
    const size_t MN4 = ((size_t)M_ROWS * NDIM) >> 2;
    const int STEP4  = (BATCH * NDIM) >> 2;                  // 4096 float4/step

    float4 s = *(const float4*)(s0 + e);

    const float4* kb = (const float4*)kvq + (e >> 2);
    const float4* vb = kb + MN4;
    const float4* qb = vb + MN4;
    float4* ob = (float4*)out + (e >> 2);

    for (int t = 0; t < T_STEPS; ++t) {
        const size_t off = (size_t)t * STEP4;
        float4 k = kb[off];
        float4 v = vb[off];
        float4 q = qb[off];
        s.x = tanhf(fmaf(s.x, 1.0f - k.x * k.x, v.x * k.x));
        s.y = tanhf(fmaf(s.y, 1.0f - k.y * k.y, v.y * k.y));
        s.z = tanhf(fmaf(s.z, 1.0f - k.z * k.z, v.z * k.z));
        s.w = tanhf(fmaf(s.w, 1.0f - k.w * k.w, v.w * k.w));
        float4 o = {s.x * q.x, s.y * q.y, s.z * q.z, s.w * q.w};
        ob[off] = o;
    }
    *((float4*)(out + (size_t)T_STEPS * BATCH * NDIM) + (e >> 2)) = s;
}

// ---------------------------------------------------------------------------
extern "C" void kernel_launch(void* const* d_in, const int* in_sizes, int n_in,
                              void* d_out, int out_size, void* d_ws, size_t ws_size,
                              hipStream_t stream) {
    (void)in_sizes; (void)n_in; (void)out_size; (void)ws_size;
    const float* x  = (const float*)d_in[0];   // [T,B,D]
    const float* s0 = (const float*)d_in[1];   // [B,N]
    const float* Wk = (const float*)d_in[2];   // [N,D]
    const float* Wv = (const float*)d_in[3];
    const float* Wq = (const float*)d_in[4];
    float* out = (float*)d_out;

    unsigned short* xb = (unsigned short*)d_ws;
    unsigned short* wb = xb + XB_ELEMS;
    float* kvq = (float*)(wb + WB_ELEMS);

    // 1) convert inputs to bf16
    cvt_bf16_kernel<<<2048, 256, 0, stream>>>(x, xb, (int)(XB_ELEMS / 4));
    cvt_bf16_kernel<<<128, 256, 0, stream>>>(Wk, wb + 0 * (size_t)NDIM * DDIM,
                                             (NDIM * DDIM) / 4);
    cvt_bf16_kernel<<<128, 256, 0, stream>>>(Wv, wb + 1 * (size_t)NDIM * DDIM,
                                             (NDIM * DDIM) / 4);
    cvt_bf16_kernel<<<128, 256, 0, stream>>>(Wq, wb + 2 * (size_t)NDIM * DDIM,
                                             (NDIM * DDIM) / 4);

    // 2) fused WMMA GEMM (TDM double-buffered LDS staging) for k|v|q
    dim3 grid(NTOT / 128, M_ROWS / 128);
    gemm_bf16_kernel<<<grid, 128, 0, stream>>>((const bf16_t*)xb,
                                               (const bf16_t*)wb, kvq);

    // 3) diagonal tanh scan (16384 chains, 4 per thread)
    scan_kernel<<<(BATCH * NDIM / 4) / 128, 128, 0, stream>>>(kvq, s0, out);
}
